// MultiHeadAttention_61546881352366
// MI455X (gfx1250) — matrix operand order repaired and checked by reading, not compiled, not run on hardware
//
#include <hip/hip_runtime.h>

// ---------------------------------------------------------------------------
// out = (v @ Wv_flat^T) @ W0^T + b0  ==  v @ (W0 @ Wv_flat)^T + b0
// Fold weights once (1024^3 bf16 WMMA GEMM), then one 16384x1024x1024
// bf16-WMMA GEMM with f32 accumulation and fused bias.
// Global->LDS movement uses the CDNA5 Tensor Data Mover: every wave issues
// TDM descriptors for its own 16-row slice of the A/B tiles of buffer
// (cur^1) while all 8 waves run WMMA on buffer (cur); per-wave TENSORcnt
// ordering + the workgroup barrier guarantee residency. The last K-step issues no prefetch.
// ---------------------------------------------------------------------------

typedef __bf16 bf16;
typedef __attribute__((ext_vector_type(16))) __bf16 v16bf;
typedef __attribute__((ext_vector_type(8)))  __bf16 v8bf;
typedef __attribute__((ext_vector_type(4)))  __bf16 v4bf;
typedef __attribute__((ext_vector_type(8)))  float  v8f;
typedef __attribute__((ext_vector_type(4)))  float  v4f;
typedef __attribute__((ext_vector_type(4)))  unsigned int v4u;
typedef __attribute__((ext_vector_type(8)))  int v8i;
typedef __attribute__((ext_vector_type(4)))  int v4i;

static constexpr int KD   = 1024;      // inner dim (== C) for both GEMMs
static constexpr int ND   = 1024;      // output cols for both GEMMs
static constexpr int BM   = 128, BN = 128, BK = 64;
static constexpr int LDSS = BK + 8;    // 72 bf16 = 144 B row stride (36 banks)
static constexpr int SLICE_ROWS  = 16;                  // rows per wave's TDM op
static constexpr int SLICE_BYTES = SLICE_ROWS * LDSS * 2;   // 2304 B
static constexpr int TILE_BYTES  = BM * LDSS * 2;       // 18432 B per tile
static constexpr int BUF_BYTES   = 2 * TILE_BYTES;      // A + B per buffer
static constexpr int LDS_TOTAL   = 2 * BUF_BYTES;       // double buffered: 72 KB

#if defined(__has_builtin)
#  if __has_builtin(__builtin_amdgcn_tensor_load_to_lds)
#    define HAVE_TDM 1
#  endif
#endif
#ifndef HAVE_TDM
#  define HAVE_TDM 0
#endif
#if __has_include(<hip/amd_detail/amd_gfx1250_TDM.h>)
#  define TDM_6ARG 1
#else
#  define TDM_6ARG 0
#endif

// ---------------- f32 -> bf16 elementwise convert (vectorized x4) ----------
// Non-temporal loads: source is read exactly once; keep L2 for bf16 data.
__global__ void cvt_f32_to_bf16(const float* __restrict__ src,
                                bf16* __restrict__ dst, int n) {
    int i = (blockIdx.x * blockDim.x + threadIdx.x) * 4;
    if (i + 3 < n) {
        v4f f = __builtin_nontemporal_load((const v4f*)(src + i));
        v4bf o;
        o[0] = (bf16)f[0]; o[1] = (bf16)f[1]; o[2] = (bf16)f[2]; o[3] = (bf16)f[3];
        *(v4bf*)(dst + i) = o;
    }
}

// ---------------- f32 [1024][1024] -> bf16 transpose (LDS-tiled) -----------
__global__ void transpose_cvt_bf16(const float* __restrict__ src,
                                   bf16* __restrict__ dst) {
    __shared__ float tile[32][33];
    const int j0 = blockIdx.x * 32;
    const int k0 = blockIdx.y * 32;
    const int tx = threadIdx.x;       // 32
    const int ty = threadIdx.y;       // 8
    #pragma unroll
    for (int r = ty; r < 32; r += 8)
        tile[r][tx] = __builtin_nontemporal_load(&src[(size_t)(k0 + r) * 1024 + j0 + tx]);
    __syncthreads();
    #pragma unroll
    for (int r = ty; r < 32; r += 8)
        dst[(size_t)(j0 + r) * 1024 + k0 + tx] = (bf16)tile[tx][r];
}

#if HAVE_TDM
// Issue one TDM 2-D tile load: `rows` x 64 bf16 from row-major [*][1024]
// into LDS at lds_off, padded 64->72 elements per row (pad_interval=4: 32
// DWORDs, pad_amount=3: 4 DWORDs).  D# packing per CDNA5 ISA ch.8.
__device__ __forceinline__ void tdm_load_slice(unsigned lds_off,
                                               const bf16* gbase,
                                               unsigned row0, unsigned col0,
                                               unsigned tensor_rows) {
    const unsigned long long ga =
        (unsigned long long)(uintptr_t)gbase +
        ((unsigned long long)row0 * (unsigned)KD + col0) * 2ull;
    v4u g0;
    g0[0] = 1u;                                   // count=1, user descriptor
    g0[1] = lds_off;                              // lds_addr (bytes)
    g0[2] = (unsigned)ga;                         // global_addr[31:0]
    g0[3] = (unsigned)(ga >> 32) | (2u << 30);    // global_addr[56:32] | type=2
    v8i g1;
    g1[0] = (int)((1u << 16)        // data_size = 1 (2 bytes)
                | (1u << 20)        // pad_enable
                | (4u << 22)        // pad_interval: 32 DWORDs (one 64-elem row)
                | (3u << 25));      // pad_amount: 4 DWORDs (8 bf16)
    g1[1] = (int)(((unsigned)KD & 0xffffu) << 16);          // tensor_dim0[15:0]
    g1[2] = (int)(((unsigned)KD >> 16) |
                  ((tensor_rows & 0xffffu) << 16));          // dim0 hi | dim1 lo
    g1[3] = (int)((tensor_rows >> 16) | ((unsigned)BK << 16)); // dim1 hi | tile_dim0
    g1[4] = (int)(unsigned)SLICE_ROWS;                       // tile_dim1 | tile_dim2=0
    g1[5] = (int)(unsigned)KD;                               // tensor_dim0_stride lo
    g1[6] = 0;                                               // stride hi | dim1_stride lo
    g1[7] = 0;
    const v4i z = {0, 0, 0, 0};
#if TDM_6ARG
    const v8i z8 = {0, 0, 0, 0, 0, 0, 0, 0};
    __builtin_amdgcn_tensor_load_to_lds(g0, g1, z, z, z8, 0);
#else
    __builtin_amdgcn_tensor_load_to_lds(g0, g1, z, z, 0);
#endif
}
#endif

// ---------------- bf16 WMMA GEMM:  C[m][n] = sum_k A[m][k] * Bt[n][k] ------
// 256 threads = 8 waves; block tile 128x128, K-step 64, TDM double buffering.
// Wave grid 2(M) x 4(N); each wave owns 4x2 WMMA tiles (16 WMMA per K-step).
template <bool OUT_BF16, bool BIAS>
__global__ __launch_bounds__(256)
void wmma_gemm_bf16(const bf16* __restrict__ A, const bf16* __restrict__ Bt,
                    float* __restrict__ Cf, bf16* __restrict__ Cb,
                    const float* __restrict__ bias, int Mrows) {
    extern __shared__ char smem[];            // LDS offset 0 (no static shared)

    const int tid   = threadIdx.x;
    const int lane  = tid & 31;
    const int wave  = tid >> 5;               // 0..7
    const int waveM = wave & 1;               // 64-row slice
    const int waveN = wave >> 1;              // 32-col slice
    const int lr    = lane & 15;
    const int hi16  = lane >> 4;
    const int khA   = hi16 * 8;               // A frag K base (interleaved layout)
    const int ksB   = hi16 * 8;               // B frag K base (interleaved layout)

    const int m0 = blockIdx.y * BM;
    const int n0 = blockIdx.x * BN;

    v8f acc[4][2];
    #pragma unroll
    for (int mi = 0; mi < 4; ++mi)
        #pragma unroll
        for (int ni = 0; ni < 2; ++ni)
            acc[mi][ni] = (v8f)0.0f;

    constexpr int NIT = KD / BK;              // 16 K-steps
    int cur = 0;

#if HAVE_TDM
    // Per-wave slice: wave w DMAs rows [w*16, w*16+16) of both tiles.
    const unsigned sliceLds = (unsigned)wave * SLICE_BYTES;
    const unsigned sliceRow = (unsigned)wave * SLICE_ROWS;
    // Prologue: fill buffer 0 (every wave issues its own two descriptors).
    tdm_load_slice(sliceLds,              A,  (unsigned)m0 + sliceRow, 0u,
                   (unsigned)Mrows);
    tdm_load_slice(sliceLds + TILE_BYTES, Bt, (unsigned)n0 + sliceRow, 0u,
                   (unsigned)ND);
#endif

    for (int it = 0; it < NIT; ++it) {
#if HAVE_TDM
        if (it + 1 < NIT) {                   // prefetch buffer cur^1
            const unsigned nb  = (unsigned)(cur ^ 1) * BUF_BYTES;
            const unsigned kb1 = (unsigned)(it + 1) * BK;
            tdm_load_slice(nb + sliceLds,              A,
                           (unsigned)m0 + sliceRow, kb1, (unsigned)Mrows);
            tdm_load_slice(nb + sliceLds + TILE_BYTES, Bt,
                           (unsigned)n0 + sliceRow, kb1, (unsigned)ND);
            __builtin_amdgcn_s_wait_tensorcnt(2);  // this wave's cur slices done
        } else {
            __builtin_amdgcn_s_wait_tensorcnt(0);
        }
#else
        // Fallback: cooperative synchronous copy into buffer `cur`.
        {
            const int kb = it * BK;
            bf16* as = (bf16*)(smem + cur * BUF_BYTES);
            bf16* bs = (bf16*)(smem + cur * BUF_BYTES + TILE_BYTES);
            #pragma unroll
            for (int u = 0; u < 4; ++u) {
                const int c    = tid * 4 + u;     // 1024 chunks of 8 bf16
                const int row  = c >> 3;
                const int kcol = (c & 7) * 8;
                *(v8bf*)(&as[row * LDSS + kcol]) =
                    *(const v8bf*)(&A[(size_t)(m0 + row) * KD + kb + kcol]);
                *(v8bf*)(&bs[row * LDSS + kcol]) =
                    *(const v8bf*)(&Bt[(size_t)(n0 + row) * KD + kb + kcol]);
            }
        }
#endif
        __syncthreads();                      // all slices of `cur` ready

        const bf16* as = (const bf16*)(smem + cur * BUF_BYTES);
        const bf16* bs = (const bf16*)(smem + cur * BUF_BYTES + TILE_BYTES);

        #pragma unroll
        for (int ks = 0; ks < BK; ks += 32) {
            // A frags: 16-bit A 16x32 layout (lanes 0-15: K {0..7,16..23}).
            v16bf af[4];
            #pragma unroll
            for (int mi = 0; mi < 4; ++mi) {
                const bf16* p = &as[(waveM * 64 + mi * 16 + lr) * LDSS + ks + khA];
                v8bf lo = *(const v8bf*)p;
                v8bf hv = *(const v8bf*)(p + 16);
                af[mi] = __builtin_shufflevector(lo, hv,
                    0,1,2,3,4,5,6,7,8,9,10,11,12,13,14,15);
            }
            // B frags: 32x16, lanes 0-15: K {0..7,16..23}, lanes 16-31: K {8..15,24..31}.
            v16bf bfg[2];
            #pragma unroll
            for (int ni = 0; ni < 2; ++ni) {
                const bf16* p = &bs[(waveN * 32 + ni * 16 + lr) * LDSS + ks + ksB];
                v8bf lo = *(const v8bf*)p;
                v8bf hv = *(const v8bf*)(p + 16);
                bfg[ni] = __builtin_shufflevector(lo, hv,
                    0,1,2,3,4,5,6,7,8,9,10,11,12,13,14,15);
            }
            #pragma unroll
            for (int mi = 0; mi < 4; ++mi)
                #pragma unroll
                for (int ni = 0; ni < 2; ++ni)
                    acc[mi][ni] = __builtin_amdgcn_wmma_f32_16x16x32_bf16(
                        false, af[mi], false, bfg[ni],
                        (short)0, acc[mi][ni], false, false);
        }
        __syncthreads();                      // done reading `cur`
        cur ^= 1;
    }

    // Epilogue: C/D layout — VGPR r: lanes 0-15 -> (M=r, N=lane),
    // lanes 16-31 -> (M=r+8, N=lane-16).  Fused bias; f32 path streams NT.
    const int rsel = hi16 * 8;
    #pragma unroll
    for (int mi = 0; mi < 4; ++mi) {
        #pragma unroll
        for (int ni = 0; ni < 2; ++ni) {
            const int gm = m0 + waveM * 64 + mi * 16 + rsel;
            const int gn = n0 + waveN * 32 + ni * 16 + lr;
            const float bv = BIAS ? bias[gn] : 0.0f;
            #pragma unroll
            for (int r = 0; r < 8; ++r) {
                const float val = acc[mi][ni][r] + bv;
                if (OUT_BF16) Cb[(size_t)(gm + r) * ND + gn] = (bf16)val;
                else __builtin_nontemporal_store(val, &Cf[(size_t)(gm + r) * ND + gn]);
            }
        }
    }
}

// ---------------------------------------------------------------------------
extern "C" void kernel_launch(void* const* d_in, const int* in_sizes, int n_in,
                              void* d_out, int out_size, void* d_ws, size_t ws_size,
                              hipStream_t stream) {
    // setup_inputs order: k, q, v, Wv, W0, b0  (k, q are dead code)
    const float* v  = (const float*)d_in[2];   // [16384, 1024]
    const float* Wv = (const float*)d_in[3];   // [1024, 1024] (H*D x C)
    const float* W0 = (const float*)d_in[4];   // [1024, 1024]
    const float* b0 = (const float*)d_in[5];   // [1024]
    float* out = (float*)d_out;                // [16384, 1024]

    constexpr size_t MV = 16384;
    char* ws = (char*)d_ws;
    bf16* vbf  = (bf16*)ws;                                   // 32 MiB
    bf16* W0bf = (bf16*)(ws + MV * 1024 * sizeof(bf16));      //  2 MiB
    bf16* WvT  = W0bf + 1024 * 1024;                          //  2 MiB
    bf16* Wf   = WvT  + 1024 * 1024;                          //  2 MiB

    // 1) convert operands to bf16 (vbf = 32 MiB, stays L2-resident)
    cvt_f32_to_bf16<<<(MV * 1024 / 4 + 255) / 256, 256, 0, stream>>>(v, vbf, MV * 1024);
    cvt_f32_to_bf16<<<(1024 * 1024 / 4 + 255) / 256, 256, 0, stream>>>(W0, W0bf, 1024 * 1024);
    transpose_cvt_bf16<<<dim3(32, 32), dim3(32, 8), 0, stream>>>(Wv, WvT);

    // 2) fold weights: Wf[i][j] = sum_k W0[i][k] * Wv[k][j]   (1024^3, WMMA)
    wmma_gemm_bf16<true, false><<<dim3(8, 8), 256, LDS_TOTAL, stream>>>(
        W0bf, WvT, nullptr, Wf, nullptr, 1024);

    // 3) main GEMM: out[m][n] = sum_k v[m][k] * Wf[n][k] + b0[n]
    wmma_gemm_bf16<false, true><<<dim3(8, 128), 256, LDS_TOTAL, stream>>>(
        vbf, Wf, out, nullptr, b0, (int)MV);
}
